// KANNeuron_60464549593379
// MI455X (gfx1250) — matrix operand order repaired and checked by reading, not compiled, not run on hardware
//
#include <hip/hip_runtime.h>

typedef __attribute__((ext_vector_type(16))) _Float16 v16h;
typedef __attribute__((ext_vector_type(2)))  __fp16   v2hp;  // cvt_pkrtz return type
typedef __attribute__((ext_vector_type(8)))  float    v8f;

#define BATCH  16384
#define DIMS   1024
#define DEGP1  9                   // reference layout: 9 terms per dim (T0..T8)
#define TERMS  8                   // T0 dropped (constant folded) -> 8 terms/dim
#define KTOT   (DIMS * TERMS)      // 8192 flattened contraction length
#define DBLK   32                  // dims processed per step
#define KBLK   (DBLK * TERMS)      // 256 local K per step = 8 WMMA chunks
#define CHUNKS (KBLK / 32)         // 8
#define ROWS   16                  // batch rows per wave (WMMA M)
#define WPB    4                   // waves per block

__launch_bounds__(128, 1)
__global__ void kan_cheb_wmma(const float* __restrict__ x,
                              const float* __restrict__ coef,
                              const float* __restrict__ hw,
                              float* __restrict__ out)
{
    // 16 KB coefficient tile (f16, k=1..8 only) + 4 x 8 KB feature tiles = 48 KB
    __shared__ __align__(16) _Float16 s_coef[KTOT];
    __shared__ __align__(16) _Float16 s_feat[WPB][ROWS * KBLK];

    const int tid  = threadIdx.x;
    const int wave = tid >> 5;
    const int lane = tid & 31;
    const int half = lane >> 4;   // 0: lanes 0-15, 1: lanes 16-31
    const int m    = lane & 15;

    // Stage coefficients for k=1..8 as f16 (flatten order d*8 + (k-1)).
    for (int i = tid; i < KTOT; i += blockDim.x) {
        int dd = i >> 3, kk = (i & 7) + 1;
        s_coef[i] = (_Float16)coef[dd * DEGP1 + kk];
    }
    __syncthreads();

    // T0 == 1 contribution: C0 = sum_d coef[d*9]  (batch-independent constant).
    // Deterministic: fixed lane-strided partials + fixed shuffle tree.
    float c0 = 0.0f;
    for (int dd = lane; dd < DIMS; dd += 32)
        c0 += coef[dd * DEGP1];
    #pragma unroll
    for (int off = 16; off > 0; off >>= 1)
        c0 += __shfl_down(c0, off, 32);
    c0 = __shfl(c0, 0, 32);

    const int b0 = (blockIdx.x * WPB + wave) * ROWS;
    _Float16* feat = &s_feat[wave][0];

    v8f acc = {};  // 16x16 f32 accumulator; all 16 columns identical (B replicated)

    for (int d0 = 0; d0 < DIMS; d0 += DBLK) {
        // ---- Phase 1: features T1..T8 for dims [d0,d0+32), rows [b0,b0+16) ----
        // lane = dim -> 32 consecutive floats per load instruction (coalesced).
        const int d = d0 + lane;
        #pragma unroll 4
        for (int r = 0; r < ROWS; ++r) {
            const float* xp = &x[(size_t)(b0 + r) * DIMS + d];
            float xv = *xp;
            __builtin_prefetch(xp + DBLK, 0, 1);   // speculative global_prefetch_b8
#if __has_builtin(__builtin_amdgcn_tanhf)
            float t = __builtin_amdgcn_tanhf(xv);  // hw v_tanh_f32
#else
            float t = tanhf(xv);
#endif
            float t2 = t + t;
            float T1 = t;
            float T2 = __builtin_fmaf(t2, T1, -1.0f);
            float T3 = __builtin_fmaf(t2, T2, -T1);
            float T4 = __builtin_fmaf(t2, T3, -T2);
            float T5 = __builtin_fmaf(t2, T4, -T3);
            float T6 = __builtin_fmaf(t2, T5, -T4);
            float T7 = __builtin_fmaf(t2, T6, -T5);
            float T8 = __builtin_fmaf(t2, T7, -T6);
            union { v2hp p[4]; uint4 q; } f;
            f.p[0] = __builtin_amdgcn_cvt_pkrtz(T1, T2);  // v_cvt_pk_f16_f32
            f.p[1] = __builtin_amdgcn_cvt_pkrtz(T3, T4);
            f.p[2] = __builtin_amdgcn_cvt_pkrtz(T5, T6);
            f.p[3] = __builtin_amdgcn_cvt_pkrtz(T7, T8);
            // one aligned 16B DS store per element (stride 16B -> no odd strides)
            *(uint4*)&feat[(r << 8) + (lane << 3)] = f.q;
        }
        // Same-wave LDS RAW: DS ops are in-order per wave; compiler inserts waits.

        // ---- Phase 2: load ALL fragments first (one dscnt wait), then 8
        //      back-to-back v_wmma_f32_16x16x32_f16 (D->C accumulate chain). ----
        union FR { unsigned int u[8]; v16h h; };
        FR fa[CHUNKS], fb[CHUNKS];
        #pragma unroll
        for (int c = 0; c < CHUNKS; ++c) {
            const int kb = c << 5;
            #pragma unroll
            for (int v = 0; v < 8; ++v) {
                // A fragment (16x32 f16): lanes 0-15 K 0-7/16-23, lanes 16-31 K 8-15/24-31
                int ka = kb + ((v >> 2) << 4) + (half << 3) + ((v & 3) << 1);
                fa[c].u[v] = *(const unsigned int*)&feat[(m << 8) + ka];
                // B fragment (32x16 f16), coefficients replicated to all 16 columns:
                // lanes 0-15 hold K 0-7/16-23, lanes 16-31 hold K 8-15/24-31.
                int kc = kb + ((v >> 2) << 4) + (half << 3) + ((v & 3) << 1);
                fb[c].u[v] = *(const unsigned int*)&s_coef[(d0 << 3) + kc];
            }
        }
        #pragma unroll
        for (int c = 0; c < CHUNKS; ++c)
            acc = __builtin_amdgcn_wmma_f32_16x16x32_f16(
                false, fa[c].h, false, fb[c].h, (short)0, acc, false, false);
    }

    // ---- Phase 3: every column of acc equals the row sum (B replicated).
    // VGPR r: lanes 0-15 -> M=r, lanes 16-31 -> M=8+r.
    const float w = hw[0];
    if (m == 0) {
        #pragma unroll
        for (int r = 0; r < 8; ++r)
            out[b0 + half * 8 + r] = w * (c0 + acc[r]);
    }
}

extern "C" void kernel_launch(void* const* d_in, const int* in_sizes, int n_in,
                              void* d_out, int out_size, void* d_ws, size_t ws_size,
                              hipStream_t stream) {
    const float* x    = (const float*)d_in[0];   // [16384, 1024] f32
    const float* coef = (const float*)d_in[1];   // [9216] f32
    const float* hw   = (const float*)d_in[2];   // [1] f32
    // d_in[3] = degree (int), fixed at 8 in this build.
    float* out = (float*)d_out;                  // [16384] f32

    dim3 grid(BATCH / (ROWS * WPB));             // 256 blocks
    dim3 block(32 * WPB);                        // 128 threads = 4 waves
    kan_cheb_wmma<<<grid, block, 0, stream>>>(x, coef, hw, out);
}
